// MultiHeadAttention_5360119185598
// MI455X (gfx1250) — matrix operand order repaired and checked by reading, not compiled, not run on hardware
//
#include <hip/hip_runtime.h>
#include <hip/hip_bf16.h>
#include <stdint.h>

typedef __bf16 bf16;
typedef __attribute__((ext_vector_type(16))) __bf16 v16bf;
typedef __attribute__((ext_vector_type(8)))  float  v8f;

#define DEVFN __device__ __forceinline__

constexpr int Bc = 4, Tc = 2048, Ec = 1024, Hc = 16, Dc = 64;
constexpr float NEG_INF = -1e30f;

// ---------------- WMMA helpers (CDNA5 wave32, 16x16x32 bf16 -> f32) ----------

DEVFN v8f wmma_bf16(v16bf a, v16bf b, v8f c) {
  // (neg_a, A, neg_b, B, c_mod, C, reuse_a, reuse_b)
  return __builtin_amdgcn_wmma_f32_16x16x32_bf16(false, a, false, b, (short)0, c,
                                                 false, false);
}

// A-matrix 16x32 (16-bit) per ISA: lane row = lane&15; g = lane>>4;
// VGPR j holds K pair k = (j<4 ? 2j : 8+2j) + 8g
DEVFN v16bf load_frag_a(const bf16* __restrict__ src, int ld) {
  const int lane = threadIdx.x & 31;
  const int row = lane & 15, g = lane >> 4;
  const bf16* r = src + (size_t)row * ld;
  v16bf a;
#pragma unroll
  for (int j = 0; j < 8; ++j) {
    const int k = ((j < 4) ? 2 * j : 8 + 2 * j) + 8 * g;
    a[2 * j] = r[k];
    a[2 * j + 1] = r[k + 1];
  }
  return a;
}

// B-matrix 32x16 (16-bit): lane column n = lane&15; g = lane>>4;
// VGPR j holds K pair k = (j<4 ? 2j : 8+2j) + 8g. srcT laid out so element (k,n) = srcT[n*ld + k]
DEVFN v16bf load_frag_b(const bf16* __restrict__ srcT, int ld) {
  const int lane = threadIdx.x & 31;
  const int n = lane & 15, g = lane >> 4;
  const bf16* r = srcT + (size_t)n * ld;
  v16bf b;
#pragma unroll
  for (int j = 0; j < 8; ++j) {
    const int k = ((j < 4) ? 2 * j : 8 + 2 * j) + 8 * g;
    b[2 * j] = r[k];
    b[2 * j + 1] = r[k + 1];
  }
  return b;
}

// ---------------- Stage 1: casts / weight repacking --------------------------

__global__ void __launch_bounds__(256) cast_f2bf(const float* __restrict__ in,
                                                 bf16* __restrict__ out, int n) {
  int i = blockIdx.x * 256 + threadIdx.x;
  if (i < n) out[i] = (bf16)in[i];
}

// W (H,E,D) fp32 -> WT (H,D,E) bf16
__global__ void __launch_bounds__(256) repack_head_w(const float* __restrict__ in,
                                                     bf16* __restrict__ out) {
  int i = blockIdx.x * 256 + threadIdx.x;  // over H*E*D
  int h = i / (Ec * Dc);
  int r = i % (Ec * Dc);
  int e = r / Dc, d = r % Dc;
  out[((size_t)h * Dc + d) * Ec + e] = (bf16)in[i];
}

// Wo (E_in, E_out) fp32 -> WoT (E_out, E_in) bf16
__global__ void __launch_bounds__(256) repack_wo(const float* __restrict__ in,
                                                 bf16* __restrict__ out) {
  int i = blockIdx.x * 256 + threadIdx.x;  // over E*E
  int k = i / Ec, n = i % Ec;
  out[(size_t)n * Ec + k] = (bf16)in[i];
}

// ---------------- shared GEMM core: 32(M) x 64(N) tile per wave --------------
// Ping-pong double-buffered K loop (unrolled x2, E multiple of 64): loads for
// the next 32-wide K-step are issued before the 8 WMMAs of the current step,
// with no register-copy rotation. acc[rt*4+nt].

DEVFN void gemm_32x64_body(const bf16* __restrict__ A, const bf16* __restrict__ W,
                           int lane, v8f acc[8]) {
  v16bf a0c = load_frag_a(A, Ec);
  v16bf a1c = load_frag_a(A + (size_t)16 * Ec, Ec);
  v16bf bc[4];
#pragma unroll
  for (int nt = 0; nt < 4; ++nt) bc[nt] = load_frag_b(W + (size_t)(nt * 16) * Ec, Ec);

  v16bf a0n, a1n, bn[4];
  for (int k0 = 0; k0 < Ec; k0 += 64) {
    {  // stage k0+32 (always in range: Ec % 64 == 0)
      const int kk = k0 + 32;
      a0n = load_frag_a(A + kk, Ec);
      a1n = load_frag_a(A + (size_t)16 * Ec + kk, Ec);
#pragma unroll
      for (int nt = 0; nt < 4; ++nt)
        bn[nt] = load_frag_b(W + (size_t)(nt * 16) * Ec + kk, Ec);
    }
    if (k0 + 96 < Ec) {  // near-cache prefetch of the weight panel ahead
      __builtin_prefetch(W + (size_t)lane * Ec + k0 + 96, 0, 3);
      __builtin_prefetch(W + (size_t)(lane + 32) * Ec + k0 + 96, 0, 3);
    }
#pragma unroll
    for (int nt = 0; nt < 4; ++nt) {
      acc[nt] = wmma_bf16(a0c, bc[nt], acc[nt]);
      acc[4 + nt] = wmma_bf16(a1c, bc[nt], acc[4 + nt]);
    }
    if (k0 + 64 < Ec) {  // stage k0+64
      const int kk = k0 + 64;
      a0c = load_frag_a(A + kk, Ec);
      a1c = load_frag_a(A + (size_t)16 * Ec + kk, Ec);
#pragma unroll
      for (int nt = 0; nt < 4; ++nt)
        bc[nt] = load_frag_b(W + (size_t)(nt * 16) * Ec + kk, Ec);
    }
#pragma unroll
    for (int nt = 0; nt < 4; ++nt) {
      acc[nt] = wmma_bf16(a0n, bn[nt], acc[nt]);
      acc[4 + nt] = wmma_bf16(a1n, bn[nt], acc[4 + nt]);
    }
  }
}

// ---------------- Stage 2: per-head Q/K/V projections ------------------------
// grid: (T/256, 3*H, B), block 256 (8 waves). Wave computes 32 rows x 64 cols.
// q,k stored (B,H,T,D); v stored transposed (B,H,D,T) for PV B-fragments.

__global__ void __launch_bounds__(256) qkv_kernel(
    const bf16* __restrict__ xb, const bf16* __restrict__ wqT,
    const bf16* __restrict__ wkT, const bf16* __restrict__ wvT,
    bf16* __restrict__ qo, bf16* __restrict__ ko, bf16* __restrict__ vto) {
  const int wave = threadIdx.x >> 5;
  const int lane = threadIdx.x & 31;
  const int n = lane & 15, g = lane >> 4;
  const int t0 = blockIdx.x * 256 + wave * 32;
  const int sel = blockIdx.y / Hc, h = blockIdx.y % Hc;
  const int b = blockIdx.z;

  const bf16* W = (sel == 0 ? wqT : (sel == 1 ? wkT : wvT)) + (size_t)h * Dc * Ec;
  const bf16* A = xb + ((size_t)b * Tc + t0) * Ec;

  v8f zero = {};
  v8f acc[8] = {zero, zero, zero, zero, zero, zero, zero, zero};
  gemm_32x64_body(A, W, lane, acc);

  const size_t bh = (size_t)b * Hc + h;
  if (sel < 2) {
    bf16* out = (sel == 0 ? qo : ko) + (bh * Tc + t0) * Dc;
#pragma unroll
    for (int rt = 0; rt < 2; ++rt)
#pragma unroll
      for (int nt = 0; nt < 4; ++nt)
#pragma unroll
        for (int i = 0; i < 8; ++i)
          out[(size_t)(rt * 16 + i + 8 * g) * Dc + nt * 16 + n] =
              (bf16)acc[rt * 4 + nt][i];
  } else {
    bf16* out = vto + bh * Dc * Tc;
#pragma unroll
    for (int rt = 0; rt < 2; ++rt)
#pragma unroll
      for (int nt = 0; nt < 4; ++nt)
#pragma unroll
        for (int i = 0; i < 8; ++i)
          out[(size_t)(nt * 16 + n) * Tc + t0 + rt * 16 + i + 8 * g] =
              (bf16)acc[rt * 4 + nt][i];
  }
}

// ---------------- Stage 3: causal flash attention ----------------------------
// grid: (T/128, H, B), block 256 (8 waves). Wave owns a 16-row Q tile, streams
// 32-wide K/V tiles with online softmax. Pipeline order per tile:
// score WMMAs -> issue V fragment loads -> softmax (overlaps V latency) ->
// LDS P relayout -> PV WMMAs. Next K/V tile prefetched into near caches.

__global__ void __launch_bounds__(256) attn_kernel(
    const bf16* __restrict__ q, const bf16* __restrict__ k,
    const bf16* __restrict__ vt, bf16* __restrict__ ob) {
  __shared__ bf16 lds_p[8][16 * 32];

  const int wave = threadIdx.x >> 5;
  const int lane = threadIdx.x & 31;
  const int n = lane & 15, g = lane >> 4;
  const int t0 = blockIdx.x * 128 + wave * 16;
  const int h = blockIdx.y, b = blockIdx.z;
  const size_t bh = (size_t)b * Hc + h;

  const bf16* Q = q + (bh * Tc + t0) * Dc;
  const bf16* K = k + bh * Tc * Dc;
  const bf16* VT = vt + bh * Dc * Tc;

  v16bf qa0 = load_frag_a(Q, Dc);
  v16bf qa1 = load_frag_a(Q + 32, Dc);

  v8f zero = {};
  v8f o[4] = {zero, zero, zero, zero};
  float m[8], l[8];
#pragma unroll
  for (int i = 0; i < 8; ++i) { m[i] = NEG_INF; l[i] = 0.f; }

  const float scale = 0.125f;  // 1/sqrt(64)
  const int send = t0 + 16;    // last needed column + 1 (causal)

  for (int s0 = 0; s0 < send; s0 += 32) {
    // ---- scores: S[16 x 32] = Q(16x64) . K^T, two 16-col subtiles ----
    v8f sc0 = zero, sc1 = zero;
    {
      v16bf kb0 = load_frag_b(K + (size_t)s0 * Dc, Dc);
      v16bf kb1 = load_frag_b(K + (size_t)s0 * Dc + 32, Dc);
      sc0 = wmma_bf16(qa0, kb0, sc0);
      sc0 = wmma_bf16(qa1, kb1, sc0);
      v16bf kb2 = load_frag_b(K + (size_t)(s0 + 16) * Dc, Dc);
      v16bf kb3 = load_frag_b(K + (size_t)(s0 + 16) * Dc + 32, Dc);
      sc1 = wmma_bf16(qa0, kb2, sc1);
      sc1 = wmma_bf16(qa1, kb3, sc1);
    }

    // ---- issue V fragment loads now; softmax VALU below hides their latency ----
    v16bf vb[4];
#pragma unroll
    for (int nt = 0; nt < 4; ++nt)
      vb[nt] = load_frag_b(VT + (size_t)(nt * 16) * Tc + s0, Tc);

    // ---- prefetch next K/V tiles into near caches (one 128B line per lane) ----
    if (s0 + 32 < send) {
      __builtin_prefetch(K + (size_t)(s0 + 32 + lane) * Dc, 0, 3);
      __builtin_prefetch(VT + (size_t)lane * Tc + s0 + 32, 0, 3);
      __builtin_prefetch(VT + (size_t)(lane + 32) * Tc + s0 + 32, 0, 3);
    }

    // ---- causal mask + online softmax (rows in VGPR idx, cols in lanes) ----
    float alpha[8];
#pragma unroll
    for (int i = 0; i < 8; ++i) {
      const int row = t0 + i + 8 * g;
      float v0 = (s0 + n <= row) ? sc0[i] * scale : NEG_INF;
      float v1 = (s0 + 16 + n <= row) ? sc1[i] * scale : NEG_INF;
      float mx = fmaxf(v0, v1);
#pragma unroll
      for (int off = 1; off < 16; off <<= 1)
        mx = fmaxf(mx, __shfl_xor(mx, off, 16));
      const float mn = fmaxf(m[i], mx);
      alpha[i] = __expf(m[i] - mn);
      const float p0 = __expf(v0 - mn);
      const float p1 = __expf(v1 - mn);
      float s = p0 + p1;
#pragma unroll
      for (int off = 1; off < 16; off <<= 1)
        s += __shfl_xor(s, off, 16);
      l[i] = l[i] * alpha[i] + s;
      m[i] = mn;
      const int rl = i + 8 * g;
      lds_p[wave][rl * 32 + n] = (bf16)p0;
      lds_p[wave][rl * 32 + 16 + n] = (bf16)p1;
    }
#pragma unroll
    for (int nt = 0; nt < 4; ++nt)
#pragma unroll
      for (int i = 0; i < 8; ++i) o[nt][i] *= alpha[i];

    // wave-local LDS visibility: all ds_stores done before re-reading.
    // (LDS ops of one wave issue and complete in order, so no extra WAR wait
    // is needed before next iteration's stores.)
    asm volatile("s_wait_dscnt 0" ::: "memory");

    // ---- reload P as A-fragment (16x32) ----
    v16bf pa;
    {
      const int rr = lane & 15;
#pragma unroll
      for (int j = 0; j < 8; ++j) {
        const int kk = ((j < 4) ? 2 * j : 8 + 2 * j) + 8 * g;
        pa[2 * j] = lds_p[wave][rr * 32 + kk];
        pa[2 * j + 1] = lds_p[wave][rr * 32 + kk + 1];
      }
    }

    // ---- O += P(16x32) . V(32x64) ----
#pragma unroll
    for (int nt = 0; nt < 4; ++nt) o[nt] = wmma_bf16(pa, vb[nt], o[nt]);
  }

  // ---- epilogue: normalize, write concat-head layout (B, T, H*D) bf16 ----
  bf16* OB = ob + ((size_t)b * Tc + t0) * (Hc * Dc) + h * Dc;
#pragma unroll
  for (int nt = 0; nt < 4; ++nt)
#pragma unroll
    for (int i = 0; i < 8; ++i) {
      const float val = o[nt][i] / l[i];
      OB[(size_t)(i + 8 * g) * (Hc * Dc) + nt * 16 + n] = (bf16)val;
    }
}

// ---------------- Stage 4: output projection + bias (fp32 out) ---------------
// grid: ((B*T)/256, E/64), block 256. Wave computes 32 rows x 64 cols.

__global__ void __launch_bounds__(256) proj_kernel(
    const bf16* __restrict__ ob, const bf16* __restrict__ woT,
    const float* __restrict__ bo, float* __restrict__ out) {
  const int wave = threadIdx.x >> 5;
  const int lane = threadIdx.x & 31;
  const int n = lane & 15, g = lane >> 4;
  const int r0 = blockIdx.x * 256 + wave * 32;
  const int n0 = blockIdx.y * 64;

  const bf16* A = ob + (size_t)r0 * Ec;
  const bf16* W = woT + (size_t)n0 * Ec;

  v8f zero = {};
  v8f acc[8] = {zero, zero, zero, zero, zero, zero, zero, zero};
  gemm_32x64_body(A, W, lane, acc);

#pragma unroll
  for (int rt = 0; rt < 2; ++rt)
#pragma unroll
    for (int nt = 0; nt < 4; ++nt)
#pragma unroll
      for (int i = 0; i < 8; ++i) {
        const int col = n0 + nt * 16 + n;
        out[(size_t)(r0 + rt * 16 + i + 8 * g) * Ec + col] =
            acc[rt * 4 + nt][i] + bo[col];
      }
}

// ---------------- host side --------------------------------------------------

extern "C" void kernel_launch(void* const* d_in, const int* in_sizes, int n_in,
                              void* d_out, int out_size, void* d_ws, size_t ws_size,
                              hipStream_t stream) {
  (void)in_sizes; (void)n_in; (void)out_size; (void)ws_size;
  const float* x  = (const float*)d_in[0];
  const float* Wq = (const float*)d_in[1];
  const float* Wk = (const float*)d_in[2];
  const float* Wv = (const float*)d_in[3];
  const float* Wo = (const float*)d_in[4];
  const float* bo = (const float*)d_in[5];
  float* out = (float*)d_out;

  char* ws = (char*)d_ws;
  constexpr size_t NX = (size_t)Bc * Tc * Ec;        // 8,388,608
  constexpr size_t NW = (size_t)Hc * Ec * Dc;        // 1,048,576
  constexpr size_t OFF_XB = 0;
  constexpr size_t OFF_WQ = OFF_XB + NX * 2;
  constexpr size_t OFF_WK = OFF_WQ + NW * 2;
  constexpr size_t OFF_WV = OFF_WK + NW * 2;
  constexpr size_t OFF_WO = OFF_WV + NW * 2;
  constexpr size_t OFF_Q  = OFF_WO + (size_t)Ec * Ec * 2;
  constexpr size_t OFF_K  = OFF_Q + NX * 2;
  constexpr size_t OFF_VT = OFF_K + NX * 2;
  constexpr size_t OFF_OB = OFF_VT + NX * 2;

  bf16* xb  = (bf16*)(ws + OFF_XB);
  bf16* wqT = (bf16*)(ws + OFF_WQ);
  bf16* wkT = (bf16*)(ws + OFF_WK);
  bf16* wvT = (bf16*)(ws + OFF_WV);
  bf16* woT = (bf16*)(ws + OFF_WO);
  bf16* qb  = (bf16*)(ws + OFF_Q);
  bf16* kb  = (bf16*)(ws + OFF_K);
  bf16* vtb = (bf16*)(ws + OFF_VT);
  bf16* obb = (bf16*)(ws + OFF_OB);

  // Stage 1: casts / repacks
  cast_f2bf<<<(int)(NX / 256), 256, 0, stream>>>(x, xb, (int)NX);
  repack_head_w<<<(int)(NW / 256), 256, 0, stream>>>(Wq, wqT);
  repack_head_w<<<(int)(NW / 256), 256, 0, stream>>>(Wk, wkT);
  repack_head_w<<<(int)(NW / 256), 256, 0, stream>>>(Wv, wvT);
  repack_wo<<<(Ec * Ec) / 256, 256, 0, stream>>>(Wo, woT);

  // Stage 2: Q/K/V projections (32x64 tiles/wave, ping-pong double-buffered)
  dim3 g2(Tc / 256, 3 * Hc, Bc);
  qkv_kernel<<<g2, 256, 0, stream>>>(xb, wqT, wkT, wvT, qb, kb, vtb);

  // Stage 3: causal flash attention
  dim3 g3(Tc / 128, Hc, Bc);
  attn_kernel<<<g3, 256, 0, stream>>>(qb, kb, vtb, obb);

  // Stage 4: output projection + bias
  dim3 g4((Bc * Tc) / 256, Ec / 64);
  proj_kernel<<<g4, 256, 0, stream>>>(obb, woT, bo, out);
}
